// MLP_Diag_20083267076267
// MI455X (gfx1250) — matrix operand order repaired and checked by reading, not compiled, not run on hardware
//
#include <hip/hip_runtime.h>
#include <hip/hip_bf16.h>
#include <math.h>

#define NN   10000
#define DD   512
#define TOPK 21          // k+1
#define IVAL 6.0f
#define EXPM6 0.00247875218f   // elu(0*6-6)+1 = exp(-6)

typedef __attribute__((ext_vector_type(16))) _Float16 v16h;
typedef __attribute__((ext_vector_type(8)))  _Float16 v8h;
typedef __attribute__((ext_vector_type(8)))  float    v8f;

// ---------------------------------------------------------------------------
// Kernel 1: h = normalize(relu(f*w0)*w1) per row, emit f16 (fits in L2: 10 MB)
// ---------------------------------------------------------------------------
__global__ __launch_bounds__(128) void row_norm_f16(
    const float* __restrict__ f, const float* __restrict__ w0,
    const float* __restrict__ w1, _Float16* __restrict__ hf) {
  const int row = blockIdx.x;
  const int t   = threadIdx.x;
  const float* fr = f + (size_t)row * DD;
  float y[4];
  float ss = 0.f;
#pragma unroll
  for (int j = 0; j < 4; ++j) {
    int d = t + j * 128;
    float x = fr[d] * w0[d];
    x = fmaxf(x, 0.f);
    x *= w1[d];
    y[j] = x;
    ss += x * x;
  }
  __shared__ float red[128];
  red[t] = ss;
  __syncthreads();
  for (int s = 64; s > 0; s >>= 1) {
    if (t < s) red[t] += red[t + s];
    __syncthreads();
  }
  const float scale = 1.f / fmaxf(sqrtf(red[0]), 1e-12f);
  _Float16* hr = hf + (size_t)row * DD;
#pragma unroll
  for (int j = 0; j < 4; ++j) {
    int d = t + j * 128;
    hr[d] = (_Float16)(y[j] * scale);
  }
}

// ---------------------------------------------------------------------------
// Kernel 2: sim = h * h^T via v_wmma_f32_16x16x32_f16.
// Wave tile 32x32 (4 WMMA tiles), 4 waves/block -> 64x64 block tile.
// A frag (16x32 f16, ISA 7.12.2): lane<16 : K[kk..kk+7] & K[kk+16..kk+23]
//                                 lane>=16: K[kk+8..kk+15] & K[kk+24..kk+31]
// B frag (32x16 f16): lane&15 = column, same K split as A frag
// ---------------------------------------------------------------------------
__device__ __forceinline__ void store_ctile(float* __restrict__ out, v8f c,
                                            int mbase, int nbase, int l15, int hi) {
  int n = nbase + l15;
  if (n >= NN) return;
#pragma unroll
  for (int r = 0; r < 8; ++r) {
    int m = mbase + hi * 8 + r;
    if (m < NN) out[(size_t)m * NN + n] = c[r];
  }
}

__global__ __launch_bounds__(128) void sim_gemm(const _Float16* __restrict__ hf,
                                                float* __restrict__ out) {
  const int lane = threadIdx.x & 31;
  const int wave = threadIdx.x >> 5;
  const int l15  = lane & 15;
  const int hi   = lane >> 4;

  const int row0 = blockIdx.y * 64 + (wave >> 1) * 32;
  const int col0 = blockIdx.x * 64 + (wave & 1) * 32;

  // clamp fragment source rows; out-of-range C entries are masked on store
  const _Float16* pa0 = hf + (size_t)min(row0 +      l15, NN - 1) * DD;
  const _Float16* pa1 = hf + (size_t)min(row0 + 16 + l15, NN - 1) * DD;
  const _Float16* pb0 = hf + (size_t)min(col0 +      l15, NN - 1) * DD;
  const _Float16* pb1 = hf + (size_t)min(col0 + 16 + l15, NN - 1) * DD;

  v8f c00 = {}, c01 = {}, c10 = {}, c11 = {};

  for (int kk = 0; kk < DD; kk += 32) {
    v8h a0lo = *(const v8h*)(pa0 + kk + hi * 8);
    v8h a0hi = *(const v8h*)(pa0 + kk + 16 + hi * 8);
    v8h a1lo = *(const v8h*)(pa1 + kk + hi * 8);
    v8h a1hi = *(const v8h*)(pa1 + kk + 16 + hi * 8);
    v16h a0 = __builtin_shufflevector(a0lo, a0hi, 0,1,2,3,4,5,6,7,8,9,10,11,12,13,14,15);
    v16h a1 = __builtin_shufflevector(a1lo, a1hi, 0,1,2,3,4,5,6,7,8,9,10,11,12,13,14,15);
    v16h b0 = __builtin_shufflevector(*(const v8h*)(pb0 + kk + hi * 8), *(const v8h*)(pb0 + kk + 16 + hi * 8), 0,1,2,3,4,5,6,7,8,9,10,11,12,13,14,15);
    v16h b1 = __builtin_shufflevector(*(const v8h*)(pb1 + kk + hi * 8), *(const v8h*)(pb1 + kk + 16 + hi * 8), 0,1,2,3,4,5,6,7,8,9,10,11,12,13,14,15);

    c00 = __builtin_amdgcn_wmma_f32_16x16x32_f16(false, a0, false, b0, (short)0, c00, false, false);
    c01 = __builtin_amdgcn_wmma_f32_16x16x32_f16(false, a0, false, b1, (short)0, c01, false, false);
    c10 = __builtin_amdgcn_wmma_f32_16x16x32_f16(false, a1, false, b0, (short)0, c10, false, false);
    c11 = __builtin_amdgcn_wmma_f32_16x16x32_f16(false, a1, false, b1, (short)0, c11, false, false);
  }

  store_ctile(out, c00, row0,      col0,      l15, hi);
  store_ctile(out, c01, row0,      col0 + 16, l15, hi);
  store_ctile(out, c10, row0 + 16, col0,      l15, hi);
  store_ctile(out, c11, row0 + 16, col0 + 16, l15, hi);
}

// ---------------------------------------------------------------------------
// Kernel 3: per-row 21st-largest threshold.
// Per-thread register top-21 (unrolled insert), then 21-round block argmax
// merge of the sorted per-thread lists via LDS.
// ---------------------------------------------------------------------------
__global__ __launch_bounds__(256) void row_topk_thresh(const float* __restrict__ sim,
                                                       float* __restrict__ thresh) {
  const int row = blockIdx.x;
  const int tid = threadIdx.x;
  const float* srow = sim + (size_t)row * NN;

  float t[TOPK];
#pragma unroll
  for (int i = 0; i < TOPK; ++i) t[i] = -3.4e38f;

  for (int i = tid; i < NN; i += 256) {
    float v = srow[i];
    if (v > t[TOPK - 1]) {
      t[TOPK - 1] = v;
#pragma unroll
      for (int s = TOPK - 1; s > 0; --s) {
        if (t[s] > t[s - 1]) { float tmp = t[s - 1]; t[s - 1] = t[s]; t[s] = tmp; }
      }
    }
  }

  __shared__ unsigned long long red[256];
  float kth = 0.f;
  for (int sel = 0; sel < TOPK; ++sel) {
    // monotone unsigned key for float ordering
    unsigned b   = __float_as_uint(t[0]);
    unsigned key = (b & 0x80000000u) ? ~b : (b | 0x80000000u);
    red[tid] = ((unsigned long long)key << 32) | (unsigned)tid;
    __syncthreads();
    for (int s = 128; s > 0; s >>= 1) {
      if (tid < s) { unsigned long long o = red[tid + s]; if (o > red[tid]) red[tid] = o; }
      __syncthreads();
    }
    unsigned long long win = red[0];
    __syncthreads();
    if (sel == TOPK - 1) {
      unsigned k2 = (unsigned)(win >> 32);
      unsigned bb = (k2 & 0x80000000u) ? (k2 ^ 0x80000000u) : ~k2;
      kth = __uint_as_float(bb);
    }
    if (tid == (int)(win & 0xffffffffu)) {  // winner pops its head
#pragma unroll
      for (int s = 0; s < TOPK - 1; ++s) t[s] = t[s + 1];
      t[TOPK - 1] = -3.4e38f;
    }
  }
  if (tid == 0) thresh[row] = kth;
}

// ---------------------------------------------------------------------------
// Kernel 4: fused top-k mask + elu(6x-6)+1, in place on d_out. NN%4==0 so each
// float4 stays within one row.
// ---------------------------------------------------------------------------
__global__ __launch_bounds__(256) void mask_elu(float* __restrict__ sim,
                                                const float* __restrict__ thresh) {
  const size_t total4 = (size_t)NN * NN / 4;
  size_t i4 = (size_t)blockIdx.x * 256 + threadIdx.x;
  if (i4 >= total4) return;
  const int row = (int)((i4 * 4) / NN);
  const float th = thresh[row];
  float4 v = ((const float4*)sim)[i4];
  float in[4] = {v.x, v.y, v.z, v.w};
  float r[4];
#pragma unroll
  for (int j = 0; j < 4; ++j) {
    float x = in[j];
    if (x >= th) {
      float z = IVAL * x - IVAL;
      r[j] = (z > 0.f) ? (z + 1.f) : __expf(z);
    } else {
      r[j] = EXPM6;
    }
  }
  ((float4*)sim)[i4] = make_float4(r[0], r[1], r[2], r[3]);
}

// ---------------------------------------------------------------------------
extern "C" void kernel_launch(void* const* d_in, const int* in_sizes, int n_in,
                              void* d_out, int out_size, void* d_ws, size_t ws_size,
                              hipStream_t stream) {
  const float* features = (const float*)d_in[0];
  const float* w0       = (const float*)d_in[1];
  const float* w1       = (const float*)d_in[2];
  float*       out      = (float*)d_out;

  _Float16* hf     = (_Float16*)d_ws;                                   // 10.24 MB
  float*    thresh = (float*)((char*)d_ws + (size_t)NN * DD * sizeof(_Float16)); // 40 KB

  row_norm_f16<<<NN, 128, 0, stream>>>(features, w0, w1, hf);

  dim3 g((NN + 63) / 64, (NN + 63) / 64);
  sim_gemm<<<g, 128, 0, stream>>>(hf, out);

  row_topk_thresh<<<NN, 256, 0, stream>>>(out, thresh);

  const size_t total4 = (size_t)NN * NN / 4;
  mask_elu<<<(unsigned)((total4 + 255) / 256), 256, 0, stream>>>(out, thresh);
}
